// RandLANetSegmentation_58394375356477
// MI455X (gfx1250) — matrix operand order repaired and checked by reading, not compiled, not run on hardware
//
#include <hip/hip_runtime.h>

// ---------------------------------------------------------------------------
// RandLA-Net segmentation forward for MI455X (gfx1250, wave32, WMMA).
// Channel GEMMs -> v_wmma_f32_16x16x32_f16, LDS-staged panels (f32 -> packed
// f16 via v_cvt_pk_rtz_f16_f32, ds_store_b128), branchless bounds handling,
// load/convert phases split so all 32 chunk loads are in flight before the
// first wait. W strip staged once per block; 16x128 block tile = 2 WMMA
// tiles per wave sharing one A fragment. BN is two-phase. att_pool gather
// is folded into the GEMM B-panel staging.
// ---------------------------------------------------------------------------

typedef __attribute__((ext_vector_type(16))) _Float16     f16x16;
typedef __attribute__((ext_vector_type(8)))  _Float16     f16x8;
typedef __attribute__((ext_vector_type(2)))  _Float16     f16x2;
typedef __attribute__((ext_vector_type(8)))  float        f32x8;
typedef __attribute__((ext_vector_type(2)))  unsigned int u32x2;
typedef __attribute__((ext_vector_type(4)))  unsigned int u32x4;

__device__ __forceinline__ f16x2 pack2(float a, float b) {
#if __has_builtin(__builtin_amdgcn_cvt_pkrtz)
  // builtin returns __fp16 ext_vector(2); same bits as f16x2 -> bit-cast
  typedef __attribute__((ext_vector_type(2))) __fp16 fp16v2;
  fp16v2 t = __builtin_amdgcn_cvt_pkrtz(a, b);   // v_cvt_pk_rtz_f16_f32
  f16x2 r;
  __builtin_memcpy(&r, &t, sizeof(r));
  return r;
#else
  f16x2 r; r.x = (_Float16)a; r.y = (_Float16)b; return r;
#endif
}

#define GEMM_COLS 128
#define MAX_CI    384

// ======================== WMMA GEMM =========================================
// Y[b,o,j] = sum_c W[o,c] * X[b,c,col(j)],  col(j) = colmap ? colmap[b][j] : j
// Block: 128 threads = 4 waves; block tile = 16 outputs x 128 columns,
// each wave computes two 16x16 tiles reusing one A fragment.
// LDS: W strip 16xCI (staged once), X panel 128x32 column-major-per-column,
// so fragment loads are contiguous ds_load_b128 pairs.
__global__ __launch_bounds__(128) void k_gemm(
    const float* __restrict__ W, const float* __restrict__ X,
    float* __restrict__ Y, const int* __restrict__ colmap,
    int CO, int CI, int NX, int M)
{
  __shared__ __align__(16) _Float16 Wl[16][MAX_CI];
  __shared__ __align__(16) _Float16 Xl[GEMM_COLS][32];

  const int b    = blockIdx.z;
  const int o0   = blockIdx.y << 4;
  const int j0   = blockIdx.x * GEMM_COLS;
  const int tid  = threadIdx.x;
  const int lane = tid & 31;
  const int wave = tid >> 5;
  const int hf   = lane >> 4;     // wave half
  const int mrow = lane & 15;

  const float* __restrict__ Xb = X + (size_t)b * CI * NX;
  const int*   __restrict__ cm = colmap ? (colmap + (size_t)b * M) : nullptr;

  // one staged column per thread (gather fold-in happens here)
  int src = -1;
  {
    const int j = j0 + tid;
    if (j < M) { src = cm ? cm[j] : j; if (src < 0 || src >= NX) src = -1; }
  }
  const float* __restrict__ px    = Xb + (src >= 0 ? src : 0);
  const float               smask = (src >= 0) ? 1.f : 0.f;

  // ---- stage full W strip 16 x CI once (branchless, unconditional loads) ----
  {
    const int    wo    = tid >> 3;             // row 0..15
    const int    wc    = (tid & 7) << 2;       // col base (x4)
    const float  wmask = ((o0 + wo) < CO) ? 1.f : 0.f;
    const size_t wbase = (size_t)(((o0 + wo) < CO) ? (o0 + wo) : 0) * CI;
    for (int kb = 0; kb < CI; kb += 32) {
      union { f16x2 p[2]; u32x2 q; } uw;
      #pragma unroll
      for (int i = 0; i < 2; ++i) {
        const int c0 = kb + wc + 2 * i, c1 = c0 + 1;
        const float a  = W[wbase + (c0 < CI ? c0 : 0)] * (c0 < CI ? wmask : 0.f);
        const float bv = W[wbase + (c1 < CI ? c1 : 0)] * (c1 < CI ? wmask : 0.f);
        uw.p[i] = pack2(a, bv);
      }
      *(u32x2*)&Wl[wo][kb + wc] = uw.q;        // ds_store_b64
    }
  }
  // (first __syncthreads below covers W-stage visibility)

  f32x8 acc0 = {0.f,0.f,0.f,0.f,0.f,0.f,0.f,0.f};
  f32x8 acc1 = {0.f,0.f,0.f,0.f,0.f,0.f,0.f,0.f};

  for (int kb = 0; kb < CI; kb += 32) {
    // ---- stage X column: 32 vals/thread, 4x ds_store_b128 ----
    {
      union { f16x2 p[16]; u32x4 q[4]; } ux;
      if (src >= 0 && kb + 32 <= CI) {
        // fast path: issue all 32 loads (strided running pointer), then pack
        float vbuf[32];
        const float* p0 = px + (size_t)kb * NX;
        #pragma unroll
        for (int i = 0; i < 32; ++i) { vbuf[i] = *p0; p0 += NX; }
        #pragma unroll
        for (int i = 0; i < 16; ++i) ux.p[i] = pack2(vbuf[2 * i], vbuf[2 * i + 1]);
      } else {                             // tail: clamped loads * float mask
        #pragma unroll
        for (int i = 0; i < 16; ++i) {
          const int c0 = kb + 2 * i, c1 = c0 + 1;
          const float a  = px[(size_t)(c0 < CI ? c0 : 0) * NX] * (c0 < CI ? smask : 0.f);
          const float bv = px[(size_t)(c1 < CI ? c1 : 0) * NX] * (c1 < CI ? smask : 0.f);
          ux.p[i] = pack2(a, bv);
        }
      }
      #pragma unroll
      for (int qq = 0; qq < 4; ++qq)
        *(u32x4*)&Xl[tid][qq << 3] = ux.q[qq];
    }
    if (kb + 32 < CI)  // hint next K-panel (global_prefetch_b8)
      __builtin_prefetch(px + (size_t)(kb + 32) * NX, 0, 1);
    __syncthreads();

    // A fragment: lane row = mrow, K = kb + hf*8 + [0,8) then +16
    f16x8 alo = *(const f16x8*)&Wl[mrow][kb + (hf << 3)];
    f16x8 ahi = *(const f16x8*)&Wl[mrow][kb + 16 + (hf << 3)];
    f16x16 afrag = __builtin_shufflevector(alo, ahi,
        0,1,2,3,4,5,6,7,8,9,10,11,12,13,14,15);
    // B fragments: lane col = wave*32 + mrow (+16), K = hf*8 + [0,8) then +16
    const int jc0 = (wave << 5) + mrow;
    f16x8 b0lo = *(const f16x8*)&Xl[jc0][hf << 3];
    f16x8 b0hi = *(const f16x8*)&Xl[jc0][(hf << 3) + 16];
    f16x16 bf0 = __builtin_shufflevector(b0lo, b0hi,
        0,1,2,3,4,5,6,7,8,9,10,11,12,13,14,15);
    f16x8 b1lo = *(const f16x8*)&Xl[jc0 + 16][hf << 3];
    f16x8 b1hi = *(const f16x8*)&Xl[jc0 + 16][(hf << 3) + 16];
    f16x16 bf1 = __builtin_shufflevector(b1lo, b1hi,
        0,1,2,3,4,5,6,7,8,9,10,11,12,13,14,15);

    acc0 = __builtin_amdgcn_wmma_f32_16x16x32_f16(
        false, afrag, false, bf0, (short)0, acc0, false, false);
    acc1 = __builtin_amdgcn_wmma_f32_16x16x32_f16(
        false, afrag, false, bf1, (short)0, acc1, false, false);
    __syncthreads();
  }

  // D layout: VGPR r -> row (r + hf*8), col = lane&15
  float* __restrict__ Yb = Y + (size_t)b * CO * M;
  const int jn0 = j0 + (wave << 5) + mrow;
  const int jn1 = jn0 + 16;
  #pragma unroll
  for (int r = 0; r < 8; ++r) {
    const int o = o0 + r + (hf << 3);
    if (o < CO) {
      if (jn0 < M) Yb[(size_t)o * M + jn0] = acc0[r];
      if (jn1 < M) Yb[(size_t)o * M + jn1] = acc1[r];
    }
  }
}

// ================= per-channel BN stats -> (scale, shift) ==================
__global__ __launch_bounds__(256) void k_ch_scale(
    const float* __restrict__ Y, const float* __restrict__ gam,
    const float* __restrict__ bet, float* __restrict__ scale,
    float* __restrict__ shift, int C, int M, int B)
{
  const int o = blockIdx.x;
  float s = 0.f, s2 = 0.f;
  for (int b = 0; b < B; ++b) {
    const float* p = Y + ((size_t)b * C + o) * M;
    for (int j = threadIdx.x; j < M; j += 256) {
      float v = p[j]; s += v; s2 += v * v;
    }
  }
  __shared__ float rs[256], rq[256];
  rs[threadIdx.x] = s; rq[threadIdx.x] = s2;
  __syncthreads();
  for (int w = 128; w > 0; w >>= 1) {
    if (threadIdx.x < w) {
      rs[threadIdx.x] += rs[threadIdx.x + w];
      rq[threadIdx.x] += rq[threadIdx.x + w];
    }
    __syncthreads();
  }
  if (threadIdx.x == 0) {
    const float n = (float)M * (float)B;
    const float mean = rs[0] / n;
    const float var  = rq[0] / n - mean * mean;
    const float sc   = gam[o] * rsqrtf(var + 1e-5f);
    scale[o] = sc;
    shift[o] = bet[o] - mean * sc;
  }
}

// ==================== apply BN + ReLU in place =============================
__global__ __launch_bounds__(256) void k_bn_relu(
    float* __restrict__ Y, const float* __restrict__ scale,
    const float* __restrict__ shift, int C, int M, long total)
{
  const long i = (long)blockIdx.x * 256 + threadIdx.x;
  if (i >= total) return;
  const int c = (int)((i / (long)M) % C);
  const float v = Y[i] * scale[c] + shift[c];
  Y[i] = v > 0.f ? v : 0.f;
}

// ==================== residual add + LeakyReLU(0.2) ========================
__global__ __launch_bounds__(256) void k_add_leaky(
    const float* __restrict__ A, const float* __restrict__ Bp,
    float* __restrict__ O, long total)
{
  const long i = (long)blockIdx.x * 256 + threadIdx.x;
  if (i >= total) return;
  const float v = A[i] + Bp[i];
  O[i] = v >= 0.f ? v : 0.2f * v;
}

// ==================== approximate KNN over 64 candidates ===================
// one 64-thread block (2 waves) per query point
__global__ __launch_bounds__(64) void k_knn(
    const float* __restrict__ H, const int* __restrict__ cand,
    int* __restrict__ idx, int C, int N, int NC, int K)
{
  const int b = blockIdx.y, n = blockIdx.x, t = threadIdx.x;
  const float* Hb = H + (size_t)b * C * N;
  __shared__ float d2s[64];
  __shared__ int   cidx[64];
  const int cnd = cand[((size_t)b * N + n) * NC + t];
  float d2 = 0.f;
  for (int c = 0; c < C; ++c) {
    const float d = Hb[(size_t)c * N + n] - Hb[(size_t)c * N + cnd];
    d2 += d * d;
  }
  d2s[t] = d2; cidx[t] = cnd;
  __syncthreads();
  if (t == 0) {
    int* out = idx + ((size_t)b * N + n) * K;
    for (int kk = 0; kk < K; ++kk) {
      float best = d2s[0]; int bi = 0;
      for (int i2 = 1; i2 < NC; ++i2)
        if (d2s[i2] < best) { best = d2s[i2]; bi = i2; }
      out[kk] = cidx[bi];
      d2s[bi] = 3.4e38f;
    }
  }
}

// ========== attention scores: relu(bn(S)) dot w2, + b2, softmax over K =====
__global__ __launch_bounds__(64) void k_att_scores(
    const float* __restrict__ S, const float* __restrict__ scale,
    const float* __restrict__ shift, const float* __restrict__ w2,
    const float* __restrict__ b2, float* __restrict__ scores,
    int C, int M, int N, int K)
{
  const int b = blockIdx.y, n = blockIdx.x, t = threadIdx.x;
  const int k = t & 15, part = t >> 4;   // 4 channel partitions per k
  const float* Sb = S + (size_t)b * C * M;
  const size_t col = (size_t)n * K + k;
  float acc = 0.f;
  for (int c = part; c < C; c += 4) {
    const float v = Sb[(size_t)c * M + col] * scale[c] + shift[c];
    acc += w2[c] * (v > 0.f ? v : 0.f);
  }
  __shared__ float red[64];
  __shared__ float logit[16];
  red[t] = acc;
  __syncthreads();
  if (t < 16) logit[t] = red[t] + red[t + 16] + red[t + 32] + red[t + 48] + b2[0];
  __syncthreads();
  if (t == 0) {
    float mx = logit[0];
    for (int i = 1; i < 16; ++i) mx = fmaxf(mx, logit[i]);
    float e[16]; float sum = 0.f;
    for (int i = 0; i < 16; ++i) { e[i] = __expf(logit[i] - mx); sum += e[i]; }
    const float inv = 1.f / sum;
    float* out = scores + ((size_t)b * N + n) * K;
    for (int i = 0; i < 16; ++i) out[i] = e[i] * inv;
  }
}

// ============ pooled[b,c,n] = sum_k H[b,c,idx[n,k]] * scores[b,n,k] ========
__global__ __launch_bounds__(256) void k_att_pool(
    const float* __restrict__ H, const int* __restrict__ idx,
    const float* __restrict__ scores, float* __restrict__ out,
    int C, int N, long total)
{
  const long i = (long)blockIdx.x * 256 + threadIdx.x;
  if (i >= total) return;
  const int n = (int)(i % N);
  const int c = (int)((i / N) % C);
  const int b = (int)(i / ((long)N * C));
  const float* Hb = H + ((size_t)b * C + c) * N;
  const int*   id = idx + ((size_t)b * N + n) * 16;
  const float* sc = scores + ((size_t)b * N + n) * 16;
  float acc = 0.f;
  #pragma unroll
  for (int kk = 0; kk < 16; ++kk) acc += Hb[id[kk]] * sc[kk];
  out[i] = acc;
}

// ==================== column gather (downsample) ===========================
__global__ __launch_bounds__(256) void k_gather_cols(
    const float* __restrict__ X, const int* __restrict__ ds,
    float* __restrict__ Y, int C, int Nin, int Nout, long total)
{
  const long i = (long)blockIdx.x * 256 + threadIdx.x;
  if (i >= total) return;
  const int n = (int)(i % Nout);
  const int c = (int)((i / Nout) % C);
  const int b = (int)(i / ((long)Nout * C));
  const int s = ds[(size_t)b * Nout + n];
  Y[i] = X[((size_t)b * C + c) * Nin + s];
}

// ==================== nearest upsample (low) + concat (high) ===============
__global__ __launch_bounds__(256) void k_up_concat(
    const float* __restrict__ LOW, const float* __restrict__ HIGH,
    float* __restrict__ OUT, int CL, int CH, int NL, int NH, long total)
{
  const long i = (long)blockIdx.x * 256 + threadIdx.x;
  if (i >= total) return;
  const int n = (int)(i % NH);
  const int c = (int)((i / NH) % (CL + CH));
  const int b = (int)(i / ((long)NH * (CL + CH)));
  float v;
  if (c < CL) {
    int jl = (int)(((long)n * NL) / NH);
    if (jl > NL - 1) jl = NL - 1;
    v = LOW[((size_t)b * CL + c) * NL + jl];
  } else {
    v = HIGH[((size_t)b * CH + (c - CL)) * NH + n];
  }
  OUT[i] = v;
}

// ==================== classifier bias add ==================================
__global__ __launch_bounds__(256) void k_bias(
    float* __restrict__ Y, const float* __restrict__ bias,
    int C, int M, long total)
{
  const long i = (long)blockIdx.x * 256 + threadIdx.x;
  if (i >= total) return;
  const int c = (int)((i / (long)M) % C);
  Y[i] += bias[c];
}

static inline dim3 g1(long total) {
  return dim3((unsigned)((total + 255) / 256));
}

// ===========================================================================
extern "C" void kernel_launch(void* const* d_in, const int* in_sizes, int n_in,
                              void* d_out, int out_size, void* d_ws, size_t ws_size,
                              hipStream_t stream)
{
  (void)in_sizes; (void)n_in; (void)out_size; (void)ws_size;
  const int B = 2, N1 = 16384, N2 = 4096, N3 = 1024, KNB = 16, NC = 64;

  auto F = [&](int i) { return (const float*)d_in[i]; };
  auto I = [&](int i) { return (const int*)d_in[i]; };

  const float* x     = F(0);
  const int*   cand1 = I(1);
  const int*   cand2 = I(2);
  const int*   cand3 = I(3);
  const int*   ds1   = I(4);
  const int*   ds2   = I(5);

  // params leaves in jax-pytree (alphabetical) order
  struct MlpP { const float *b, *g, *w; };
  struct ApP  { const float *b1, *b2, *g1; MlpP mlp; const float *w1, *w2; };
  struct LfaP { ApP ap1, ap2; MlpP post, pre, shortp; };
  int pi = 6;
  auto rM = [&]() { MlpP m; m.b = F(pi++); m.g = F(pi++); m.w = F(pi++); return m; };
  auto rA = [&]() { ApP a; a.b1 = F(pi++); a.b2 = F(pi++); a.g1 = F(pi++);
                    a.mlp = rM(); a.w1 = F(pi++); a.w2 = F(pi++); return a; };
  auto rL = [&]() { LfaP l; l.ap1 = rA(); l.ap2 = rA();
                    l.post = rM(); l.pre = rM(); l.shortp = rM(); return l; };

  const float* cls_b = F(pi++);     // 'cls_b'
  MlpP cls_mlp = rM();              // 'cls_mlp'
  const float* cls_w = F(pi++);     // 'cls_w'
  MlpP dec1 = rM();                 // 'dec1'
  MlpP dec2 = rM();                 // 'dec2'
  MlpP dec3 = rM();                 // 'dec3'
  MlpP enc0 = rM();                 // 'enc0'
  LfaP L1 = rL();                   // 'lfa1'
  LfaP L2 = rL();                   // 'lfa2'
  LfaP L3 = rL();                   // 'lfa3'

  // -------- workspace carve-out (floats) --------
  float* wsf = (float*)d_ws;
  size_t off = 0;
  auto A = [&](size_t n) { float* p = wsf + off; off += n; return p; };
  float* e0 = A((size_t)B * 32  * N1);
  float* e1 = A((size_t)B * 64  * N1);
  float* e2 = A((size_t)B * 128 * N2);
  float* e3 = A((size_t)B * 256 * N3);
  float* d1 = A((size_t)B * 64  * N2);
  float* d2 = A((size_t)B * 128 * N3);
  float* u2 = A((size_t)B * 128 * N2);
  float* u1 = A((size_t)B * 64  * N1);
  float* u0 = A((size_t)B * 32  * N1);
  float* hc = A((size_t)B * 32  * N1);
  float* T1 = A((size_t)B * 192 * N1);   // also decoder concat staging
  float* T2 = A((size_t)B * 192 * N1);
  float* T3 = A((size_t)B * 192 * N1);
  float* T4 = A((size_t)B * 192 * N1);
  float* SBIG = A((size_t)B * 64 * N1 * KNB);  // att_pool raw GEMM (largest level)
  float* SCR  = A((size_t)B * N1 * KNB);
  float* sc = A(512);
  float* sh = A(512);
  int* idx1 = (int*)A((size_t)B * N1 * KNB);
  int* idx2 = (int*)A((size_t)B * N2 * KNB);
  int* idx3 = (int*)A((size_t)B * N3 * KNB);

  // -------- launch helpers --------
  auto gemm = [&](const float* W, const float* X_, float* Y, const int* cm,
                  int CO, int CI, int NX, int M) {
    dim3 grid((unsigned)((M + GEMM_COLS - 1) / GEMM_COLS),
              (unsigned)((CO + 15) / 16), (unsigned)B);
    k_gemm<<<grid, 128, 0, stream>>>(W, X_, Y, cm, CO, CI, NX, M);
  };
  auto bnrelu = [&](float* Y, const float* g, const float* bt, int C, int M) {
    k_ch_scale<<<C, 256, 0, stream>>>(Y, g, bt, sc, sh, C, M, B);
    long total = (long)B * C * M;
    k_bn_relu<<<g1(total), 256, 0, stream>>>(Y, sc, sh, C, M, total);
  };
  auto mlp = [&](const MlpP& m, const float* X_, float* Y, int CO, int CI, int M) {
    gemm(m.w, X_, Y, nullptr, CO, CI, M, M);
    bnrelu(Y, m.g, m.b, CO, M);
  };
  auto att = [&](const ApP& a, const float* Hin, const int* idx, float* Hout,
                 int C, int N) {
    const int M = N * KNB;
    gemm(a.w1, Hin, SBIG, idx, C, C, N, M);                 // gather fused in B-panel
    k_ch_scale<<<C, 256, 0, stream>>>(SBIG, a.g1, a.b1, sc, sh, C, M, B);
    k_att_scores<<<dim3((unsigned)N, (unsigned)B), 64, 0, stream>>>(
        SBIG, sc, sh, a.w2, a.b2, SCR, C, M, N, KNB);
    long tp = (long)B * C * N;
    k_att_pool<<<g1(tp), 256, 0, stream>>>(Hin, idx, SCR, T4, C, N, tp);
    mlp(a.mlp, T4, Hout, C, C, N);
  };
  auto lfa = [&](const LfaP& l, const float* Xin, const int* cand, int* idx,
                 float* Eout, int CI, int CO, int N) {
    mlp(l.shortp, Xin, T1, CO, CI, N);                      // shortcut branch
    mlp(l.pre, Xin, T2, CO, CI, N);                         // h
    k_knn<<<dim3((unsigned)N, (unsigned)B), 64, 0, stream>>>(
        T2, cand, idx, CO, N, NC, KNB);
    att(l.ap1, T2, idx, T3, CO, N);
    att(l.ap2, T3, idx, T2, CO, N);
    mlp(l.post, T2, T3, CO, CO, N);
    long t = (long)B * CO * N;
    k_add_leaky<<<g1(t), 256, 0, stream>>>(T3, T1, Eout, t);
  };

  // -------- forward pass --------
  mlp(enc0, x, e0, 32, 10, N1);
  lfa(L1, e0, cand1, idx1, e1, 32, 64, N1);
  { long t = (long)B * 64 * N2;
    k_gather_cols<<<g1(t), 256, 0, stream>>>(e1, ds1, d1, 64, N1, N2, t); }
  lfa(L2, d1, cand2, idx2, e2, 64, 128, N2);
  { long t = (long)B * 128 * N3;
    k_gather_cols<<<g1(t), 256, 0, stream>>>(e2, ds2, d2, 128, N2, N3, t); }
  lfa(L3, d2, cand3, idx3, e3, 128, 256, N3);

  // decoder
  { long t = (long)B * (256 + 128) * N2;
    k_up_concat<<<g1(t), 256, 0, stream>>>(e3, e2, T1, 256, 128, N3, N2, t); }
  mlp(dec3, T1, u2, 128, 384, N2);
  { long t = (long)B * (128 + 64) * N1;
    k_up_concat<<<g1(t), 256, 0, stream>>>(u2, e1, T1, 128, 64, N2, N1, t); }
  mlp(dec2, T1, u1, 64, 192, N1);
  { long t = (long)B * (64 + 32) * N1;
    k_up_concat<<<g1(t), 256, 0, stream>>>(u1, e0, T1, 64, 32, N1, N1, t); }
  mlp(dec1, T1, u0, 32, 96, N1);
  mlp(cls_mlp, u0, hc, 32, 32, N1);

  // classifier: (2x32) GEMM + bias -> d_out (B,2,N1)
  float* out = (float*)d_out;
  gemm(cls_w, hc, out, nullptr, 2, 32, N1, N1);
  { long t = (long)B * 2 * N1;
    k_bias<<<g1(t), 256, 0, stream>>>(out, cls_b, 2, N1, t); }
}
